// U_GCN_EGNN_23055384445177
// MI455X (gfx1250) — matrix operand order repaired and checked by reading, not compiled, not run on hardware
//
#include <hip/hip_runtime.h>
#include <hip/hip_bf16.h>

typedef __attribute__((ext_vector_type(8)))  __bf16 v8bf;
typedef __attribute__((ext_vector_type(16))) __bf16 v16bf;
typedef __attribute__((ext_vector_type(8)))  float  v8f;
typedef unsigned int uint32x4 __attribute__((ext_vector_type(4)));
typedef int          int32x8  __attribute__((ext_vector_type(8)));
typedef int          int32x4  __attribute__((ext_vector_type(4)));

#define HIDF 128
#define KP1 288                    // 2*IN + EDGE_F + 1 = 259 padded to 9 k-tiles of 32

// swizzled bf16 weight sizes (halfs)
#define EW1_H (288 * 128)
#define W128_H (128 * 128)
#define NW1_H (256 * 128)
#define EDGE_W_BYTES ((EW1_H + W128_H + W128_H) * 2)   // eW1|eW2|cW1 = 139264 B
#define NODE_W_BYTES ((NW1_H + W128_H) * 2)            // nW1|nW2     =  98304 B
#define EDGE_WAVE_BYTES 13824      // feA 9216 + hid 4096 + scratch 448, padded
#define NODE_WAVE_BYTES 12288      // feA 8192 + hid 4096
#define EDGE_SMEM (EDGE_W_BYTES + 8 * EDGE_WAVE_BYTES) // 249856 (< 320KB WGP LDS)
#define NODE_SMEM (NODE_W_BYTES + 8 * NODE_WAVE_BYTES) // 196608

__device__ __forceinline__ unsigned short f2bf(float f) {
  unsigned int u = __float_as_uint(f);
  u += 0x7FFFu + ((u >> 16) & 1u);           // round-to-nearest-even
  return (unsigned short)(u >> 16);
}
__device__ __forceinline__ float bf2f(unsigned short h) {
  return __uint_as_float(((unsigned int)h) << 16);
}
__device__ __forceinline__ float silu_f(float x) {
  return x * (1.0f / (1.0f + __expf(-x)));
}
// wave-local ordering for cross-lane LDS handoffs (per-wave private staging,
// LDS ops are in-order within a wave; this stops compiler reordering)
__device__ __forceinline__ void wave_lds_fence() {
  __builtin_amdgcn_fence(__ATOMIC_ACQ_REL, "wavefront");
  __builtin_amdgcn_wave_barrier();
}

// TDM: DMA `ndwords` dwords from global -> LDS as a 1-row 2D tile (ISA ch.8 D#).
// ndwords must be <= 65535 (tile_dim0 is 16-bit). Issue from ONE wave only.
__device__ __forceinline__ void tdm_load_1d(void* lds_dst, const void* gsrc,
                                            unsigned int ndwords) {
  unsigned long long ga = (unsigned long long)(uintptr_t)gsrc;
  unsigned int la = (unsigned int)(uintptr_t)lds_dst;  // low 32 bits = LDS offset
  uint32x4 g0;
  g0[0] = 1u;                                    // count=1 (valid user D#)
  g0[1] = la;                                    // lds_addr
  g0[2] = (unsigned int)ga;                      // global_addr[31:0]
  g0[3] = (unsigned int)(ga >> 32) | (2u << 30); // global_addr[56:32] | type=2
  int32x8 g1;
  g1[0] = (int)(2u << 16);                       // wg_mask=0, data_size=4B
  g1[1] = (int)((ndwords & 0xFFFFu) << 16);      // tensor_dim0[15:0]
  g1[2] = (int)((ndwords >> 16) | (1u << 16));   // tensor_dim0[31:16] | tensor_dim1=1
  g1[3] = (int)((ndwords & 0xFFFFu) << 16);      // tile_dim0 = ndwords
  g1[4] = 1;                                     // tile_dim1=1, tile_dim2=0
  g1[5] = (int)ndwords;                          // tensor_dim0_stride[31:0]
  g1[6] = 0;
  g1[7] = 0;
  int32x4 z4 = {0, 0, 0, 0};                     // D# groups 2/3: unused (<=2D)
  int32x8 z8 = {0, 0, 0, 0, 0, 0, 0, 0};         // extended descriptor slot
  __builtin_amdgcn_tensor_load_to_lds(g0, g1, z4, z4, z8, 0);
}

// A-fragment (16x32 bf16, M=lane%16): lanes 0-15 hold K {kt*32+0..7, +16..23},
// lanes 16-31 hold K {kt*32+8..15, +24..31}  (ISA 7.12.2)
__device__ __forceinline__ v16bf load_a_frag(const unsigned short* rowbase, int kbase) {
  v8bf a0 = *reinterpret_cast<const v8bf*>(rowbase + kbase);
  v8bf a1 = *reinterpret_cast<const v8bf*>(rowbase + kbase + 16);
  return __builtin_shufflevector(a0, a1, 0,1,2,3,4,5,6,7,8,9,10,11,12,13,14,15);
}

// 16 x KP x 128 GEMM: A from per-wave LDS ([16][KP] bf16 rows), B from LDS-resident
// pre-swizzled fragments. All 8 B-fragments of a k-tile are loaded into
// independent registers first so the ds_loads clause up and the dscnt waits
// stagger instead of serializing load->wait(0)->wmma.
__device__ __forceinline__ void gemm_tile(const unsigned short* ldsA, int KP,
                                          const unsigned short* Wsw, v8f acc[8],
                                          int lane) {
  const int row  = lane & 15;
  const int kofs = (lane & 16) ? 8 : 0;
  const unsigned short* abase = ldsA + row * KP;
  const int nkt = KP >> 5;
  for (int kt = 0; kt < nkt; ++kt) {
    v16bf a = load_a_frag(abase, (kt << 5) + kofs);
    const v16bf* bp = reinterpret_cast<const v16bf*>(Wsw) + (kt << 8) + lane;
    v16bf b[8];
#pragma unroll
    for (int nt = 0; nt < 8; ++nt) b[nt] = bp[nt << 5];
#pragma unroll
    for (int nt = 0; nt < 8; ++nt)
      acc[nt] = __builtin_amdgcn_wmma_f32_16x16x32_bf16(
          false, a, false, b[nt], (short)0, acc[nt], false, false);
  }
}

// Pre-swizzle W[Kreal x 128] (f32 row-major) -> bf16 B-fragment layout, zero-padded.
// out[((kt*8+nt)*32+lane)*16 + j] = W[kt*32 + (lane&16?8:0) + (j&7) + (j&8?16:0)][nt*16 + lane%16]
__global__ void swizzle_w_kernel(const float* __restrict__ W, unsigned short* __restrict__ out,
                                 int Kreal, int total) {
  int tid = blockIdx.x * blockDim.x + threadIdx.x;
  if (tid >= total) return;
  int j    = tid & 15;
  int lane = (tid >> 4) & 31;
  int nt   = (tid >> 9) & 7;
  int kt   = tid >> 12;
  int n = nt * 16 + (lane & 15);
  int k = (kt << 5) + ((lane & 16) ? 8 : 0) + (j & 7) + ((j & 8) ? 16 : 0);
  float v = (k < Kreal) ? W[(size_t)k * 128 + n] : 0.0f;
  out[tid] = f2bf(v);
}

__global__ void zero_kernel(float* __restrict__ p, int n) {
  int i = blockIdx.x * blockDim.x + threadIdx.x;
  if (i < n) p[i] = 0.0f;
}

__global__ void __launch_bounds__(256) egnn_edge_kernel(
    const float* __restrict__ node_feat, const float* __restrict__ coord,
    const float* __restrict__ edge_feat, const int* __restrict__ src,
    const int* __restrict__ dst,
    const unsigned short* __restrict__ edgeW,  // eW1p|eW2p|cW1p contiguous
    const float* __restrict__ eb1, const float* __restrict__ eb2,
    const float* __restrict__ cb1, const float* __restrict__ cW2,
    float* __restrict__ h_neigh, float* __restrict__ x_sum,
    float* __restrict__ deg, int nTiles, int nE) {
  extern __shared__ __align__(16) char smem[];
  unsigned short* wE1 = (unsigned short*)smem;
  unsigned short* wE2 = wE1 + EW1_H;
  unsigned short* wC1 = wE2 + W128_H;

  const int warp = threadIdx.x >> 5;
  const int lane = threadIdx.x & 31;
  char* my = smem + EDGE_W_BYTES + warp * EDGE_WAVE_BYTES;
  unsigned short* feA  = (unsigned short*)my;             // 9216 B
  unsigned short* hid  = (unsigned short*)(my + 9216);    // 4096 B
  float* s_xd  = (float*)(my + 13312);                    // 192 B
  float* s_rad = (float*)(my + 13504);                    // 64 B
  int*   s_src = (int*)(my + 13568);
  int*   s_dst = (int*)(my + 13632);
  int*   s_ok  = (int*)(my + 13696);

  // DMA the bf16-swizzled weights into LDS once per workgroup (wave 0 only)
  if (threadIdx.x < 32) {
    tdm_load_1d(wE1, edgeW, EDGE_W_BYTES >> 2);
    __builtin_amdgcn_s_wait_tensorcnt(0);
  }
  __syncthreads();

  const int nloc = lane & 15;
  const int mo   = (lane & 16) ? 8 : 0;
  const int stride = gridDim.x * 8;

  for (int t = blockIdx.x * 8 + warp; t < nTiles; t += stride) {
    const int e0 = t * 16;

    if (lane < 16) {
      int e  = e0 + lane;
      int ok = (e < nE);
      int is = ok ? src[e] : 0;
      int id = ok ? dst[e] : 0;
      s_src[lane] = is; s_dst[lane] = id; s_ok[lane] = ok;
      float dx = coord[(size_t)is*3+0] - coord[(size_t)id*3+0];
      float dy = coord[(size_t)is*3+1] - coord[(size_t)id*3+1];
      float dz = coord[(size_t)is*3+2] - coord[(size_t)id*3+2];
      float r  = dx*dx + dy*dy + dz*dz;
      float inv = 1.0f / (sqrtf(r) + 1e-30f);
      s_rad[lane] = r;
      s_xd[lane*3+0] = dx*inv; s_xd[lane*3+1] = dy*inv; s_xd[lane*3+2] = dz*inv;
    }
    wave_lds_fence();

    // gather f = [h_src | h_dst | radial | edge_feat | pad] -> bf16 LDS [16][288]
    for (int idx = lane; idx < 16 * KP1; idx += 32) {
      int e = idx / KP1;
      int k = idx - e * KP1;
      float v = 0.0f;
      if (s_ok[e]) {
        if (k < 128)       v = node_feat[(size_t)s_src[e]*128 + k];
        else if (k < 256)  v = node_feat[(size_t)s_dst[e]*128 + (k - 128)];
        else if (k == 256) v = s_rad[e];
        else if (k < 259)  v = edge_feat[(size_t)(e0 + e)*2 + (k - 257)];
      }
      feA[idx] = f2bf(v);
    }
    wave_lds_fence();

    v8f zero = {0.f,0.f,0.f,0.f,0.f,0.f,0.f,0.f};
    v8f acc[8];

    // edge MLP layer 1: [16,288] x [288,128], SiLU
#pragma unroll
    for (int nt = 0; nt < 8; ++nt) acc[nt] = zero;
    gemm_tile(feA, KP1, wE1, acc, lane);
#pragma unroll
    for (int nt = 0; nt < 8; ++nt) {
      float b = eb1[nt*16 + nloc];
#pragma unroll
      for (int v = 0; v < 8; ++v)
        hid[(v + mo) * HIDF + nt*16 + nloc] = f2bf(silu_f(acc[nt][v] + b));
    }
    wave_lds_fence();

    // edge MLP layer 2: SiLU -> msg_h; scatter-add into h_neigh[dst]
#pragma unroll
    for (int nt = 0; nt < 8; ++nt) acc[nt] = zero;
    gemm_tile(hid, HIDF, wE2, acc, lane);
    wave_lds_fence();
#pragma unroll
    for (int nt = 0; nt < 8; ++nt) {
      float b = eb2[nt*16 + nloc];
#pragma unroll
      for (int v = 0; v < 8; ++v) {
        int m = v + mo;
        float x = silu_f(acc[nt][v] + b);
        hid[m * HIDF + nt*16 + nloc] = f2bf(x);
        if (s_ok[m])
          atomicAdd(&h_neigh[(size_t)s_dst[m]*HIDF + nt*16 + nloc], x);
      }
    }
    wave_lds_fence();

    // coord MLP layer 1: SiLU
#pragma unroll
    for (int nt = 0; nt < 8; ++nt) acc[nt] = zero;
    gemm_tile(hid, HIDF, wC1, acc, lane);
    wave_lds_fence();
#pragma unroll
    for (int nt = 0; nt < 8; ++nt) {
      float b = cb1[nt*16 + nloc];
#pragma unroll
      for (int v = 0; v < 8; ++v)
        hid[(v + mo) * HIDF + nt*16 + nloc] = f2bf(silu_f(acc[nt][v] + b));
    }
    wave_lds_fence();

    // coord MLP layer 2: [128 -> 1] dot, then msg_x = s * x_diff scatter
    {
      int m  = lane & 15;
      int kb = (lane & 16) ? 64 : 0;
      float s = 0.0f;
      for (int k = 0; k < 64; ++k)
        s += bf2f(hid[m * HIDF + kb + k]) * cW2[kb + k];
      s += __shfl_xor(s, 16, 32);
      if (lane < 16 && s_ok[m]) {
        int d = s_dst[m];
        atomicAdd(&x_sum[(size_t)d*3 + 0], s * s_xd[m*3+0]);
        atomicAdd(&x_sum[(size_t)d*3 + 1], s * s_xd[m*3+1]);
        atomicAdd(&x_sum[(size_t)d*3 + 2], s * s_xd[m*3+2]);
        atomicAdd(&deg[d], 1.0f);
      }
    }
    wave_lds_fence();
  }
}

__global__ void __launch_bounds__(256) egnn_node_kernel(
    const float* __restrict__ node_feat, const float* __restrict__ coord,
    const unsigned short* __restrict__ nodeW,  // nW1p|nW2p contiguous
    const float* __restrict__ nb1, const float* __restrict__ nb2,
    const float* __restrict__ h_neigh, const float* __restrict__ x_sum,
    const float* __restrict__ deg,
    float* __restrict__ h_out, float* __restrict__ x_out,
    int nTiles, int nN) {
  extern __shared__ __align__(16) char smem[];
  unsigned short* wN1 = (unsigned short*)smem;
  unsigned short* wN2 = wN1 + NW1_H;

  const int warp = threadIdx.x >> 5;
  const int lane = threadIdx.x & 31;
  char* my = smem + NODE_W_BYTES + warp * NODE_WAVE_BYTES;
  unsigned short* feA = (unsigned short*)my;            // 8192 B
  unsigned short* hid = (unsigned short*)(my + 8192);   // 4096 B

  if (threadIdx.x < 32) {
    tdm_load_1d(wN1, nodeW, NODE_W_BYTES >> 2);
    __builtin_amdgcn_s_wait_tensorcnt(0);
  }
  __syncthreads();

  const int nloc = lane & 15;
  const int mo   = (lane & 16) ? 8 : 0;
  const int stride = gridDim.x * 8;

  for (int t = blockIdx.x * 8 + warp; t < nTiles; t += stride) {
    const int n0 = t * 16;

    // gather [node_feat | h_neigh] -> bf16 LDS [16][256]
    for (int idx = lane; idx < 16 * 256; idx += 32) {
      int r = idx >> 8;
      int k = idx & 255;
      int node = n0 + r;
      float v = 0.0f;
      if (node < nN)
        v = (k < 128) ? node_feat[(size_t)node*128 + k]
                      : h_neigh[(size_t)node*128 + (k - 128)];
      feA[idx] = f2bf(v);
    }
    wave_lds_fence();

    v8f zero = {0.f,0.f,0.f,0.f,0.f,0.f,0.f,0.f};
    v8f acc[8];

    // node MLP layer 1: [16,256] x [256,128], SiLU
#pragma unroll
    for (int nt = 0; nt < 8; ++nt) acc[nt] = zero;
    gemm_tile(feA, 256, wN1, acc, lane);
#pragma unroll
    for (int nt = 0; nt < 8; ++nt) {
      float b = nb1[nt*16 + nloc];
#pragma unroll
      for (int v = 0; v < 8; ++v)
        hid[(v + mo) * HIDF + nt*16 + nloc] = f2bf(silu_f(acc[nt][v] + b));
    }
    wave_lds_fence();

    // node MLP layer 2 (no activation) -> h_out
#pragma unroll
    for (int nt = 0; nt < 8; ++nt) acc[nt] = zero;
    gemm_tile(hid, HIDF, wN2, acc, lane);
#pragma unroll
    for (int nt = 0; nt < 8; ++nt) {
      float b = nb2[nt*16 + nloc];
#pragma unroll
      for (int v = 0; v < 8; ++v) {
        int node = n0 + v + mo;
        if (node < nN)
          h_out[(size_t)node*HIDF + nt*16 + nloc] = acc[nt][v] + b;
      }
    }

    // x = coord + x_sum / max(deg, 1)
    for (int i = lane; i < 48; i += 32) {
      int node = n0 + i / 3;
      int c = i - (i / 3) * 3;
      if (node < nN) {
        float dg = fmaxf(deg[node], 1.0f);
        x_out[(size_t)node*3 + c] =
            coord[(size_t)node*3 + c] + x_sum[(size_t)node*3 + c] / dg;
      }
    }
    wave_lds_fence();
  }
}

extern "C" void kernel_launch(void* const* d_in, const int* in_sizes, int n_in,
                              void* d_out, int out_size, void* d_ws, size_t ws_size,
                              hipStream_t stream) {
  const float* node_feat = (const float*)d_in[0];
  const float* coord     = (const float*)d_in[1];
  const float* edge_feat = (const float*)d_in[2];
  const int*   src       = (const int*)d_in[3];
  const int*   dst       = (const int*)d_in[4];
  const float* eW1 = (const float*)d_in[5];
  const float* eb1 = (const float*)d_in[6];
  const float* eW2 = (const float*)d_in[7];
  const float* eb2 = (const float*)d_in[8];
  const float* nW1 = (const float*)d_in[9];
  const float* nb1 = (const float*)d_in[10];
  const float* nW2 = (const float*)d_in[11];
  const float* nb2 = (const float*)d_in[12];
  const float* cW1 = (const float*)d_in[13];
  const float* cb1 = (const float*)d_in[14];
  const float* cW2 = (const float*)d_in[15];

  const int nN = in_sizes[0] / 128;
  const int nE = in_sizes[3];

  // workspace layout: edge weights contiguous (eW1p|eW2p|cW1p), then node
  // weights contiguous (nW1p|nW2p), then f32 scatter accumulators.
  char* ws = (char*)d_ws;
  unsigned short* eW1p = (unsigned short*)ws; ws += (size_t)EW1_H * 2;
  unsigned short* eW2p = (unsigned short*)ws; ws += (size_t)W128_H * 2;
  unsigned short* cW1p = (unsigned short*)ws; ws += (size_t)W128_H * 2;
  unsigned short* nW1p = (unsigned short*)ws; ws += (size_t)NW1_H * 2;
  unsigned short* nW2p = (unsigned short*)ws; ws += (size_t)W128_H * 2;
  float* h_neigh = (float*)ws; ws += (size_t)nN * 128 * 4;
  float* x_sum   = (float*)ws; ws += (size_t)nN * 3 * 4;
  float* deg     = (float*)ws; ws += (size_t)nN * 4;

  float* h_out = (float*)d_out;
  float* x_out = h_out + (size_t)nN * 128;

  // 1) pre-swizzle weights to bf16 WMMA B-fragment layout (deterministic each call)
  { int t = 288 * 128; swizzle_w_kernel<<<(t + 255) / 256, 256, 0, stream>>>(eW1, eW1p, 259, t); }
  { int t = 128 * 128; swizzle_w_kernel<<<(t + 255) / 256, 256, 0, stream>>>(eW2, eW2p, 128, t); }
  { int t = 128 * 128; swizzle_w_kernel<<<(t + 255) / 256, 256, 0, stream>>>(cW1, cW1p, 128, t); }
  { int t = 256 * 128; swizzle_w_kernel<<<(t + 255) / 256, 256, 0, stream>>>(nW1, nW1p, 256, t); }
  { int t = 128 * 128; swizzle_w_kernel<<<(t + 255) / 256, 256, 0, stream>>>(nW2, nW2p, 128, t); }

  // 2) zero scatter accumulators (h_neigh | x_sum | deg contiguous: nN*132 floats)
  { int t = nN * 132; zero_kernel<<<(t + 255) / 256, 256, 0, stream>>>(h_neigh, t); }

  // 3) edge pass: TDM weights->LDS, persistent waves over 16-edge tiles
  {
    int nTiles = (nE + 15) / 16;
    int blocks = 768;
    if (blocks * 8 > nTiles) blocks = (nTiles + 7) / 8;
    egnn_edge_kernel<<<blocks, 256, EDGE_SMEM, stream>>>(
        node_feat, coord, edge_feat, src, dst, eW1p, eb1, eb2, cb1, cW2,
        h_neigh, x_sum, deg, nTiles, nE);
  }

  // 4) node pass: TDM weights->LDS, persistent waves over 16-node tiles
  {
    int nTiles = (nN + 15) / 16;
    int blocks = 256;
    if (blocks * 8 > nTiles) blocks = (nTiles + 7) / 8;
    egnn_node_kernel<<<blocks, 256, NODE_SMEM, stream>>>(
        node_feat, coord, nW1p, nb1, nb2, h_neigh, x_sum, deg,
        h_out, x_out, nTiles, nN);
  }
}
